// LSTM_50809463112028
// MI455X (gfx1250) — matrix operand order repaired and checked by reading, not compiled, not run on hardware
//
#include <hip/hip_runtime.h>

#define SEQ  8192
#define HID  1024
#define G4H  4096
#define REC_BLOCKS 32

typedef __attribute__((ext_vector_type(16))) __bf16 v16bf;
typedef __attribute__((ext_vector_type(8)))  float  v8f;

struct Pair16 { uint4 a, b; };

__device__ __forceinline__ v16bf to_v16bf(uint4 a, uint4 b) {
  Pair16 p{a, b};
  return __builtin_bit_cast(v16bf, p);
}

__device__ __forceinline__ unsigned short f2bf(float f) {
  unsigned int u = __float_as_uint(f);
  u = (u + 0x7FFFu + ((u >> 16) & 1u)) >> 16;   // round-to-nearest-even
  return (unsigned short)u;
}

__device__ __forceinline__ float sigm(float x) { return 1.f / (1.f + __expf(-x)); }
__device__ __forceinline__ float tanh_fast(float x) {
  float e = __expf(2.f * x);
  return (e - 1.f) / (e + 1.f);
}

// ---------------- init: zero h/h16 + barrier counter ----------------
__global__ void k_init(float* h, unsigned short* h16, int* counter) {
  int i = blockIdx.x * blockDim.x + threadIdx.x;
  if (i < HID) { h[i] = 0.f; h16[i] = 0; }
  if (i == 0) counter[0] = 0;
}

// ---------------- f32 -> bf16 (vectorized x4) ----------------
__global__ void k_f32_to_bf16(const float* __restrict__ src,
                              unsigned short* __restrict__ dst, int n4) {
  int i = blockIdx.x * blockDim.x + threadIdx.x;
  if (i >= n4) return;
  float4 v = ((const float4*)src)[i];
  ushort4 o;
  o.x = f2bf(v.x); o.y = f2bf(v.y); o.z = f2bf(v.z); o.w = f2bf(v.w);
  ((ushort4*)dst)[i] = o;
}

// ---------------- x_gates GEMM: [SEQ,HID] x [HID,G4H] -> [SEQ,G4H] + bias ----
// Block: 256 thr (8 waves) -> 128(M) x 256(N) tile. Wave: 64x64 = 16 wmma accs.
__global__ __launch_bounds__(256) void k_gemm_xg(
    const unsigned short* __restrict__ X,     // [SEQ,HID] bf16
    const unsigned short* __restrict__ W,     // [G4H,HID] bf16 (W_ih, row-major)
    const float* __restrict__ b_ih,
    const float* __restrict__ b_hh,
    float* __restrict__ xg)                   // [SEQ,G4H] f32
{
  const int lane = threadIdx.x & 31;
  const int wave = threadIdx.x >> 5;
  const int wm = wave >> 2;                   // 0..1
  const int wn = wave & 3;                    // 0..3
  const int m0 = blockIdx.y * 128 + wm * 64;
  const int n0 = blockIdx.x * 256 + wn * 64;
  const int lh = lane & 15;
  const int lv = lane >> 4;

  v8f acc[4][4];
#pragma unroll
  for (int i = 0; i < 4; ++i)
#pragma unroll
    for (int j = 0; j < 4; ++j)
      acc[i][j] = (v8f){0.f, 0.f, 0.f, 0.f, 0.f, 0.f, 0.f, 0.f};

  for (int k0 = 0; k0 < HID; k0 += 32) {
    uint4 aq0[4], aq1[4], bq0[4], bq1[4];
#pragma unroll
    for (int st = 0; st < 4; ++st) {
      // A 16-bit 16x32: lane half = K half (contiguous K, matches B)
      const uint4* p = (const uint4*)(X + (size_t)(m0 + st * 16 + lh) * HID + k0 + lv * 16);
      aq0[st] = p[0];
      aq1[st] = p[1];
    }
#pragma unroll
    for (int sn = 0; sn < 4; ++sn) {
      // B 16-bit 32x16: lane half = K half; column n = W row n (contiguous K)
      const uint4* p = (const uint4*)(W + (size_t)(n0 + sn * 16 + lh) * HID + k0 + lv * 16);
      bq0[sn] = p[0];
      bq1[sn] = p[1];
    }
#pragma unroll
    for (int st = 0; st < 4; ++st)
#pragma unroll
      for (int sn = 0; sn < 4; ++sn)
        acc[st][sn] = __builtin_amdgcn_wmma_f32_16x16x32_bf16(
            false, to_v16bf(aq0[st], aq1[st]),
            false, to_v16bf(bq0[sn], bq1[sn]),
            (short)0, acc[st][sn], false, false);
  }

#pragma unroll
  for (int sn = 0; sn < 4; ++sn) {
    const int col = n0 + sn * 16 + lh;
    const float bias = b_ih[col] + b_hh[col];
#pragma unroll
    for (int st = 0; st < 4; ++st)
#pragma unroll
      for (int r = 0; r < 8; ++r) {
        const int row = m0 + st * 16 + r + (lv << 3);   // C layout: lanes16-31 = M+8
        xg[(size_t)row * G4H + col] = acc[st][sn][r] + bias;
      }
  }
}

// ---------------- device-wide barrier (monotonic counter) ----------------
__device__ __forceinline__ void grid_sync(int* counter, int target) {
  __threadfence();
  __syncthreads();
  if (threadIdx.x == 0) {
    atomicAdd(counter, 1);
    while (__hip_atomic_load(counter, __ATOMIC_RELAXED, __HIP_MEMORY_SCOPE_AGENT) < target)
      __builtin_amdgcn_s_sleep(1);          // 64-clk poll: low wake latency
    __threadfence();
  }
  __syncthreads();
}

// ---------------- persistent LSTM recurrence ----------------
// 32 blocks x 8 waves. Block b owns hidden units [b*32, b*32+32): its 8 waves
// compute all 4 gates for those units (wave w -> gate w>>1, unit-tile w&1),
// each wave holding its 16x1024 bf16 W_hh slice register-resident (256 VGPRs).
// Gate exchange via LDS; cell state c lives in registers of the owner threads.
// => exactly ONE device-wide barrier per timestep (h16 broadcast).
__global__ __launch_bounds__(256) void k_lstm_rec(
    const float* __restrict__ xg,             // [SEQ,G4H] (biases folded in)
    const unsigned short* __restrict__ Whh,   // [G4H,HID] bf16
    float* __restrict__ h,                    // [HID] final hidden (f32)
    unsigned short* __restrict__ h16,         // [HID] bf16 broadcast buffer
    int* counter)
{
  __shared__ float lds_g[4 * 32];             // [gate][unit_local]

  const int lane = threadIdx.x & 31;
  const int wave = threadIdx.x >> 5;
  const int lh = lane & 15;
  const int lv = lane >> 4;
  const int gate = wave >> 1;                 // 0..3  (i,f,g,o)
  const int utl  = wave & 1;                  // 0..1  (unit-tile within block)
  const int n0 = gate * HID + (blockIdx.x * 2 + utl) * 16;   // gate-column tile

  // Preload register-resident W_hh slice: B[k][n] = Whh[n0+n][k]
  uint4 wq0[32], wq1[32];
#pragma unroll
  for (int kc = 0; kc < 32; ++kc) {
    const uint4* p = (const uint4*)(Whh + (size_t)(n0 + lh) * HID + kc * 32 + lv * 16);
    wq0[kc] = p[0];
    wq1[kc] = p[1];
  }

  const bool row0 = (lh == 0);                // lanes 0 and 16 carry A-row M=0
  const uint4 z = make_uint4(0u, 0u, 0u, 0u);
  float creg = 0.f;                           // cell state, register-resident
  float hv = 0.f;
  int bar = 0;

#pragma unroll 1
  for (int t = 0; t < SEQ; ++t) {
    // ---- phase A: 16 gate pre-activations per wave via single-row WMMA GEMV
    //      8 independent accumulator chains -> dependent-WMMA chain depth 4.
    v8f acc[8];
#pragma unroll
    for (int u = 0; u < 8; ++u)
      acc[u] = (v8f){0.f, 0.f, 0.f, 0.f, 0.f, 0.f, 0.f, 0.f};
#pragma unroll
    for (int kc = 0; kc < 32; ++kc) {
      uint4 a0 = z, a1 = z;
      if (row0) {
        const uint4* hp = (const uint4*)(h16 + kc * 32 + lv * 16);
        a0 = hp[0];          // K = kc*32 + lv*16 + 0..7
        a1 = hp[1];          // K = kc*32 + lv*16 + 8..15
      }
      acc[kc & 7] = __builtin_amdgcn_wmma_f32_16x16x32_bf16(
          false, to_v16bf(a0, a1),
          false, to_v16bf(wq0[kc], wq1[kc]),
          (short)0, acc[kc & 7], false, false);
    }
    if (lane < 16) {          // D row M=0 lives in element 0, lanes 0..15
      float s = ((acc[0][0] + acc[1][0]) + (acc[2][0] + acc[3][0])) +
                ((acc[4][0] + acc[5][0]) + (acc[6][0] + acc[7][0]));
      lds_g[gate * 32 + utl * 16 + lane] = s;
    }
    __syncthreads();

    // ---- phase B: cell update for this block's 32 units (LDS gates, reg c)
    if (threadIdx.x < 32) {
      const int ul = threadIdx.x;
      const int j  = blockIdx.x * 32 + ul;
      const float* xr = xg + (size_t)t * G4H;
      float gi = lds_g[ul]          + xr[j];
      float gf = lds_g[32 + ul]     + xr[HID + j];
      float gg = lds_g[64 + ul]     + xr[2 * HID + j];
      float go = lds_g[96 + ul]     + xr[3 * HID + j];
      float iv = sigm(gi), fv = sigm(gf), gv = tanh_fast(gg), ov = sigm(go);
      creg = fv * creg + iv * gv;
      hv = ov * tanh_fast(creg);
      h16[j] = f2bf(hv);
      if (t + 1 < SEQ) {      // pull next timestep's xg row toward L2/L0
        const float* xn = xg + (size_t)(t + 1) * G4H;
        __builtin_prefetch(xn + j, 0, 1);
        __builtin_prefetch(xn + HID + j, 0, 1);
        __builtin_prefetch(xn + 2 * HID + j, 0, 1);
        __builtin_prefetch(xn + 3 * HID + j, 0, 1);
      }
    }

    // ---- single device-wide barrier per step: publish h16 to all blocks
    ++bar; grid_sync(counter, bar * REC_BLOCKS);
  }

  if (threadIdx.x < 32)
    h[blockIdx.x * 32 + threadIdx.x] = hv;    // final hidden state (f32)
}

// ---------------- final projection: out = h @ W_out.T + b_out ----------------
__global__ void k_proj(const float* __restrict__ h, const float* __restrict__ Wout,
                       const float* __restrict__ bout, float* __restrict__ out) {
  int o = blockIdx.x * blockDim.x + threadIdx.x;
  if (o >= HID) return;
  const float* wrow = Wout + (size_t)o * HID;
  float s = bout[o];
  for (int k = 0; k < HID; k += 4) {
    float4 hv = *(const float4*)(h + k);
    float4 wv = *(const float4*)(wrow + k);
    s += hv.x * wv.x + hv.y * wv.y + hv.z * wv.z + hv.w * wv.w;
  }
  out[o] = s;
}

extern "C" void kernel_launch(void* const* d_in, const int* in_sizes, int n_in,
                              void* d_out, int out_size, void* d_ws, size_t ws_size,
                              hipStream_t stream) {
  const float* input_seq = (const float*)d_in[0];
  const float* W_ih = (const float*)d_in[1];
  const float* W_hh = (const float*)d_in[2];
  const float* b_ih = (const float*)d_in[3];
  const float* b_hh = (const float*)d_in[4];
  const float* W_out = (const float*)d_in[5];
  const float* b_out = (const float*)d_in[6];
  float* out = (float*)d_out;

  // workspace layout (all 16B-aligned)
  char* ws = (char*)d_ws;
  size_t off = 0;
  float* xg             = (float*)(ws + off);          off += (size_t)SEQ * G4H * 4;  // 128 MB
  unsigned short* X16   = (unsigned short*)(ws + off); off += (size_t)SEQ * HID * 2;  // 16 MB
  unsigned short* Wih16 = (unsigned short*)(ws + off); off += (size_t)G4H * HID * 2;  // 8 MB
  unsigned short* Whh16 = (unsigned short*)(ws + off); off += (size_t)G4H * HID * 2;  // 8 MB
  float* h              = (float*)(ws + off);          off += (size_t)HID * 4;
  unsigned short* h16   = (unsigned short*)(ws + off); off += 4096;
  int* counter          = (int*)(ws + off);            off += 256;

  k_init<<<4, 256, 0, stream>>>(h, h16, counter);

  k_f32_to_bf16<<<(SEQ * HID / 4) / 256, 256, 0, stream>>>(input_seq, X16, SEQ * HID / 4);
  k_f32_to_bf16<<<(G4H * HID / 4) / 256, 256, 0, stream>>>(W_ih, Wih16, G4H * HID / 4);
  k_f32_to_bf16<<<(G4H * HID / 4) / 256, 256, 0, stream>>>(W_hh, Whh16, G4H * HID / 4);

  dim3 gg(G4H / 256, SEQ / 128);   // 16 x 64 blocks
  k_gemm_xg<<<gg, 256, 0, stream>>>(X16, Wih16, b_ih, b_hh, xg);

  k_lstm_rec<<<REC_BLOCKS, 256, 0, stream>>>(xg, Whh16, h, h16, counter);

  k_proj<<<HID / 256, 256, 0, stream>>>(h, W_out, b_out, out);
}
